// mLSTMBackendTriton_77970836291655
// MI455X (gfx1250) — matrix operand order repaired and checked by reading, not compiled, not run on hardware
//
#include <hip/hip_runtime.h>

#define TT 2048
#define KD 128
#define VD 128
#define NBH 16  // B*H

typedef __bf16 bf16_t;
typedef bf16_t v16bf __attribute__((ext_vector_type(16)));
typedef float  v8f   __attribute__((ext_vector_type(8)));
typedef unsigned int u32x4 __attribute__((ext_vector_type(4)));
typedef int i32x4 __attribute__((ext_vector_type(4)));
typedef int i32x8 __attribute__((ext_vector_type(8)));

union V16 {
  v16bf v;
  unsigned short u[16];
  uint4 q[2];
};

__device__ __forceinline__ unsigned short f2bf(float f) {
  unsigned int x = __float_as_uint(f);
  x += 0x7FFFu + ((x >> 16) & 1u);  // round-to-nearest-even
  return (unsigned short)(x >> 16);
}

__device__ __forceinline__ unsigned lds_off(const void* p) {
  // generic pointers to LDS carry the byte offset in the low 32 bits
  return (unsigned)(unsigned long long)(uintptr_t)p;
}

// Issue a 2D TDM tile load (bf16 elements) with optional LDS row padding.
// pad_interval_code: pads after 2^(c+1) dwords stored; pad_amount_code: c+1 dwords.
__device__ __forceinline__ void tdm_load_2d(const void* gaddr, unsigned ldsa,
                                            unsigned tensor_d0, unsigned tensor_d1,
                                            unsigned tile_d0, unsigned tile_d1,
                                            unsigned stride0,
                                            unsigned pad_interval_code,
                                            unsigned pad_amount_code) {
  unsigned long long ga = (unsigned long long)(uintptr_t)gaddr;
  u32x4 g0;
  g0[0] = 1u;                                   // count=1 (valid user descriptor)
  g0[1] = ldsa;                                 // lds_addr (bytes)
  g0[2] = (unsigned)(ga & 0xFFFFFFFFu);         // global_addr[31:0]
  g0[3] = (unsigned)((ga >> 32) & 0x1FFFFFFu)   // global_addr[56:32]
          | (2u << 30);                         // type=2 ("image")
  i32x8 g1;
  g1[0] = (int)((1u << 16)                      // data_size=1 -> 2 bytes
                | (1u << 20)                    // pad_enable
                | (pad_interval_code << 22)
                | (pad_amount_code << 25));
  g1[1] = (int)((tensor_d0 & 0xFFFFu) << 16);                                  // dim0[15:0]
  g1[2] = (int)(((tensor_d0 >> 16) & 0xFFFFu) | ((tensor_d1 & 0xFFFFu) << 16));// dim0 hi|dim1 lo
  g1[3] = (int)(((tensor_d1 >> 16) & 0xFFFFu) | (tile_d0 << 16));              // dim1 hi|tile0
  g1[4] = (int)(tile_d1 & 0xFFFFu);             // tile1 | tile2=0
  g1[5] = (int)stride0;                         // dim0 stride [31:0]
  g1[6] = 0;                                    // stride hi / dim1 stride lo
  g1[7] = 0;
  i32x4 z4 = {0, 0, 0, 0};                      // groups 2/3 unused (2D tensor)
  i32x8 z8 = {0, 0, 0, 0, 0, 0, 0, 0};
  __builtin_amdgcn_tensor_load_to_lds(g0, g1, z4, z4, z8, 0);
}

// ---------------------------------------------------------------------------
// Kernel 1: fc = cumsum(logsigmoid(f_gate)); g = i_gate - fc. One wave/(b,h).
// ---------------------------------------------------------------------------
__global__ __launch_bounds__(32) void prep_gates(const float* __restrict__ fg,
                                                 const float* __restrict__ ig,
                                                 float* __restrict__ fc,
                                                 float* __restrict__ g) {
  const int bh = blockIdx.x;
  const int lane = threadIdx.x;
  const float* f = fg + (size_t)bh * TT;
  const float* i = ig + (size_t)bh * TT;
  float carry = 0.0f;
  for (int t0 = 0; t0 < TT; t0 += 32) {
    float x = f[t0 + lane];
    float ls = fminf(x, 0.0f) - __logf(1.0f + __expf(-fabsf(x)));
    float s = ls;
#pragma unroll
    for (int off = 1; off < 32; off <<= 1) {
      float n = __shfl_up(s, off, 32);
      if (lane >= off) s += n;
    }
    float fcv = carry + s;
    fc[(size_t)bh * TT + t0 + lane] = fcv;
    g[(size_t)bh * TT + t0 + lane] = i[t0 + lane] - fcv;
    carry += __shfl(s, 31, 32);
  }
}

// ---------------------------------------------------------------------------
// Kernel 2: K -> bf16 [BH,T,K]; V -> bf16 transposed [BH,V,T].
// ---------------------------------------------------------------------------
__global__ __launch_bounds__(256) void prep_convert(const float* __restrict__ k,
                                                    const float* __restrict__ v,
                                                    unsigned short* __restrict__ kb,
                                                    unsigned short* __restrict__ vt) {
  const int idx = blockIdx.x * blockDim.x + threadIdx.x;
  kb[idx] = f2bf(k[idx]);
  const int t = idx & (TT - 1);
  const int vc = (idx >> 11) & (VD - 1);
  const int bh = idx >> 18;
  vt[idx] = f2bf(v[((size_t)bh * TT + t) * VD + vc]);
}

// ---------------------------------------------------------------------------
// Kernel 3: flash-style chunked mLSTM forward.
//  - 4 waves/block, one 16-row query tile per wave, block covers 64 rows.
//  - K/V^T tiles double-buffered in LDS, filled by the Tensor Data Mover
//    (TDM row padding de-conflicts LDS banks), consumed via ds_load_b128.
//  - Both GEMMs on v_wmma_f32_16x16x32_bf16; online row-max stabilization.
// ---------------------------------------------------------------------------
__global__ __launch_bounds__(128) void mlstm_fwd(
    const float* __restrict__ q, const float* __restrict__ fc,
    const float* __restrict__ g, const unsigned short* __restrict__ kb,
    const unsigned short* __restrict__ vt, float* __restrict__ out) {
  // K tile: 32 rows x 128 bf16, LDS row stride 136 (64 dw + 4 dw pad)
  // V tile: 128 rows x 32 bf16, LDS row stride 40  (16 dw + 4 dw pad)
  alignas(16) __shared__ unsigned short kbuf[2][32 * 136];
  alignas(16) __shared__ unsigned short vbuf[2][128 * 40];
  alignas(16) __shared__ unsigned short sbuf[4][16 * 40];

  const int lane = threadIdx.x & 31;
  const int wave = threadIdx.x >> 5;
  const int col = lane & 15;
  const int half = lane >> 4;
  const int bh = blockIdx.y;
  const int tq = (blockIdx.x * 4 + wave) * 16;
  unsigned short* slds = sbuf[wave];

  const float scale = 0.08838834764831845f;  // 128^-0.5, folded into Q

  // ---- Q A-fragments: 4 x (16x32 bf16) ----
  V16 qa[4];
  {
    const float* qrow = q + ((size_t)bh * TT + (tq + col)) * KD;
#pragma unroll
    for (int f = 0; f < 4; ++f) {
      const float* p0 = qrow + f * 32 + half * 8;
      const float* p1 = p0 + 16;
      float4 c0 = *(const float4*)(p0);
      float4 c1 = *(const float4*)(p0 + 4);
      float4 c2 = *(const float4*)(p1);
      float4 c3 = *(const float4*)(p1 + 4);
      qa[f].u[0] = f2bf(c0.x * scale); qa[f].u[1] = f2bf(c0.y * scale);
      qa[f].u[2] = f2bf(c0.z * scale); qa[f].u[3] = f2bf(c0.w * scale);
      qa[f].u[4] = f2bf(c1.x * scale); qa[f].u[5] = f2bf(c1.y * scale);
      qa[f].u[6] = f2bf(c1.z * scale); qa[f].u[7] = f2bf(c1.w * scale);
      qa[f].u[8]  = f2bf(c2.x * scale); qa[f].u[9]  = f2bf(c2.y * scale);
      qa[f].u[10] = f2bf(c2.z * scale); qa[f].u[11] = f2bf(c2.w * scale);
      qa[f].u[12] = f2bf(c3.x * scale); qa[f].u[13] = f2bf(c3.y * scale);
      qa[f].u[14] = f2bf(c3.z * scale); qa[f].u[15] = f2bf(c3.w * scale);
    }
  }

  float fcq[8];
#pragma unroll
  for (int r = 0; r < 8; ++r) fcq[r] = fc[(size_t)bh * TT + tq + r + 8 * half];

  float m_run[8], sum_s[8];
#pragma unroll
  for (int r = 0; r < 8; ++r) { m_run[r] = -__builtin_inff(); sum_s[r] = 0.0f; }
  v8f hacc[8];
#pragma unroll
  for (int nt = 0; nt < 8; ++nt) hacc[nt] = (v8f){};

  // Block-uniform step count over the block's 64 causal rows; per-row masks
  // turn the (<=1) over-shot tiles of low-tq waves into exact no-ops.
  const int nsteps = 2 * blockIdx.x + 2;
  const unsigned short* kbase = kb + (size_t)bh * TT * KD;
  const unsigned short* vbase = vt + (size_t)bh * VD * TT;

  // Prologue: TDM-fill buffer 0 (wave 0 issues; TENSORcnt + barrier releases).
  if (wave == 0) {
    tdm_load_2d(kbase, lds_off(kbuf[0]), 128, 32, 128, 32, 128, 5, 3);
    tdm_load_2d(vbase, lds_off(vbuf[0]), 32, 128, 32, 128, TT, 3, 3);
    __builtin_amdgcn_s_wait_tensorcnt(0);
  }
  __syncthreads();

  for (int i = 0; i < nsteps; ++i) {
    const int s0 = i * 32;
    const int cur = i & 1;
    const unsigned short* kt = kbuf[cur];
    const unsigned short* vtl = vbuf[cur];

    // Async prefetch of the next K/V tiles into the other buffer.
    if (wave == 0 && (i + 1) < nsteps) {
      const int s1 = s0 + 32;
      tdm_load_2d(kbase + (size_t)s1 * KD, lds_off(kbuf[cur ^ 1]),
                  128, 32, 128, 32, 128, 5, 3);
      tdm_load_2d(vbase + s1, lds_off(vbuf[cur ^ 1]),
                  32, 128, 32, 128, TT, 3, 3);
    }

    // ---- Q * K^T: two 16x16 subtiles (8 WMMAs), K frags from LDS ----
    v8f sacc[2];
#pragma unroll
    for (int n = 0; n < 2; ++n) {
      v8f acc = (v8f){};
      const unsigned short* krow = kt + (n * 16 + col) * 136 + half * 8;
#pragma unroll
      for (int f = 0; f < 4; ++f) {
        V16 bfrag;
        bfrag.q[0] = *(const uint4*)(krow + f * 32);
        bfrag.q[1] = *(const uint4*)(krow + f * 32 + 16);
        acc = __builtin_amdgcn_wmma_f32_16x16x32_bf16(
            false, qa[f].v, false, bfrag.v, (short)0, acc, false, false);
      }
      sacc[n] = acc;
    }

    // ---- decay gating + online stabilization ----
    const float gc0 = g[(size_t)bh * TT + s0 + col];
    const float gc1 = g[(size_t)bh * TT + s0 + 16 + col];
    float scl[8];
#pragma unroll
    for (int r = 0; r < 8; ++r) {
      const int t = tq + r + 8 * half;
      const bool v0 = (s0 + col) <= t;
      const bool v1 = (s0 + 16 + col) <= t;
      const float ld0 = fcq[r] + gc0;
      const float ld1 = fcq[r] + gc1;
      float tm = fmaxf(v0 ? ld0 : -__builtin_inff(),
                       v1 ? ld1 : -__builtin_inff());
#pragma unroll
      for (int off = 1; off < 16; off <<= 1)
        tm = fmaxf(tm, __shfl_xor(tm, off, 32));  // reduce within 16-lane half
      const float nm = fmaxf(m_run[r], tm);       // fully-masked tile: nm=m_run
      scl[r] = __expf(m_run[r] - nm);             // -> 1.0, no-op rescale
      m_run[r] = nm;
      const float p0 = v0 ? __expf(ld0 - nm) : 0.0f;
      const float p1 = v1 ? __expf(ld1 - nm) : 0.0f;
      const float g0 = sacc[0][r] * p0;
      const float g1 = sacc[1][r] * p1;
      float rs = g0 + g1;
#pragma unroll
      for (int off = 1; off < 16; off <<= 1) rs += __shfl_xor(rs, off, 32);
      sum_s[r] = sum_s[r] * scl[r] + rs;
      const int M = r + 8 * half;
      slds[M * 40 + col] = f2bf(g0);
      slds[M * 40 + 16 + col] = f2bf(g1);
    }

    // rescale running output accumulators
#pragma unroll
    for (int nt = 0; nt < 8; ++nt)
#pragma unroll
      for (int r = 0; r < 8; ++r) hacc[nt][r] *= scl[r];

    // reload gated 16x32 tile as WMMA A fragment (per-wave LDS, DS in-order)
    V16 aS;
    aS.q[0] = *(const uint4*)(slds + col * 40 + half * 8);
    aS.q[1] = *(const uint4*)(slds + col * 40 + 16 + half * 8);

    // ---- S * V across 128 output columns (8 WMMAs), V frags from LDS ----
#pragma unroll
    for (int nt = 0; nt < 8; ++nt) {
      const unsigned short* vrow = vtl + (nt * 16 + col) * 40 + half * 8;
      V16 bv;
      bv.q[0] = *(const uint4*)(vrow);
      bv.q[1] = *(const uint4*)(vrow + 16);
      hacc[nt] = __builtin_amdgcn_wmma_f32_16x16x32_bf16(
          false, aS.v, false, bv.v, (short)0, hacc[nt], false, false);
    }

    // Drain the prefetch, then block barrier: next iteration may consume the
    // other buffer and (one iteration later) overwrite this one.
    if (wave == 0 && (i + 1) < nsteps) __builtin_amdgcn_s_wait_tensorcnt(0);
    __syncthreads();
  }

  // ---- epilogue: h / max(|rowsum|, exp(-m)) ----
  float inv[8];
#pragma unroll
  for (int r = 0; r < 8; ++r) {
    const float norm = fmaxf(fabsf(sum_s[r]), __expf(-m_run[r]));
    inv[r] = 1.0f / norm;
  }
#pragma unroll
  for (int nt = 0; nt < 8; ++nt) {
#pragma unroll
    for (int r = 0; r < 8; ++r) {
      const int t = tq + r + 8 * half;
      out[((size_t)bh * TT + t) * VD + nt * 16 + col] = hacc[nt][r] * inv[r];
    }
  }
}

extern "C" void kernel_launch(void* const* d_in, const int* in_sizes, int n_in,
                              void* d_out, int out_size, void* d_ws, size_t ws_size,
                              hipStream_t stream) {
  const float* q  = (const float*)d_in[0];
  const float* k  = (const float*)d_in[1];
  const float* v  = (const float*)d_in[2];
  const float* ig = (const float*)d_in[3];
  const float* fg = (const float*)d_in[4];
  float* out = (float*)d_out;

  // workspace: fc (128KB) | g (128KB) | K bf16 (8MB) | V^T bf16 (8MB)
  char* ws = (char*)d_ws;
  float* fc = (float*)ws;
  float* g  = fc + (size_t)NBH * TT;
  unsigned short* kb = (unsigned short*)(g + (size_t)NBH * TT);
  unsigned short* vt = kb + (size_t)NBH * TT * KD;

  prep_gates<<<dim3(NBH), dim3(32), 0, stream>>>(fg, ig, fc, g);

  const int nconv = NBH * TT * KD;  // 4,194,304
  prep_convert<<<dim3(nconv / 256), dim3(256), 0, stream>>>(k, v, kb, vt);

  mlstm_fwd<<<dim3(TT / 64, NBH), dim3(128), 0, stream>>>(q, fc, g, kb, vt, out);
}
